// TransformerEncoderBlock2_5497558139620
// MI455X (gfx1250) — matrix operand order repaired and checked by reading, not compiled, not run on hardware
//
#include <hip/hip_runtime.h>
#include <hip/hip_bf16.h>
#include <stdint.h>

// ---------------------------------------------------------------------------
// Problem constants
// ---------------------------------------------------------------------------
constexpr int Bsz  = 4;
constexpr int Tseq = 1024;
constexpr int Dm   = 768;
constexpr int Hh   = 12;
constexpr int DH   = 64;      // head dim
constexpr int MLPD = 3072;
constexpr int BT   = Bsz * Tseq;   // 4096 token rows
constexpr int D3   = 3 * Dm;       // 2304

typedef __bf16 bf16_t;
typedef __bf16 v16bf __attribute__((ext_vector_type(16)));
typedef __bf16 v8bf  __attribute__((ext_vector_type(8)));
typedef float  v8f   __attribute__((ext_vector_type(8)));
typedef unsigned int u32x4 __attribute__((ext_vector_type(4)));
typedef int   i32x8 __attribute__((ext_vector_type(8)));
typedef int   i32x4 __attribute__((ext_vector_type(4)));

static __device__ __forceinline__ v16bf cat16(v8bf lo, v8bf hi) {
  return __builtin_shufflevector(lo, hi, 0,1,2,3,4,5,6,7,8,9,10,11,12,13,14,15);
}
static __device__ __forceinline__ v8f vzero8() {
  v8f z = {0.f,0.f,0.f,0.f,0.f,0.f,0.f,0.f};
  return z;
}
static __device__ __forceinline__ v8f wmma_bf16(v16bf a, v16bf b, v8f c) {
  // D = A x B + C ; 16x16x32 bf16 -> f32
  return __builtin_amdgcn_wmma_f32_16x16x32_bf16(
      /*neg_a=*/false, a, /*neg_b=*/false, b,
      /*c_mod=*/(short)0, c, /*reuse_a=*/false, /*reuse_b=*/false);
}

// LDS byte offset of a __shared__ object (generic -> addrspace(3) -> offset)
static __device__ __forceinline__ uint32_t lds_off(const void* p) {
  return (uint32_t)(uintptr_t)(__attribute__((address_space(3))) const void*)p;
}

// ---------------------------------------------------------------------------
// TDM: 2-D tile load Global -> LDS (CDNA5 ISA).
// data_size = 2 bytes (bf16). LDS padding: +4 DWORDs every 16 DWORDs, i.e.
// a 32-bf16 (64B) tile row lands on a 40-bf16 (80B) LDS pitch.
// tensor dims are set equal to tile dims (tiles are always fully in-bounds).
// Toolchain here is the 6-arg builtin (clang-23 / therock-10.0 headers).
// ---------------------------------------------------------------------------
static __device__ __forceinline__ void tdm_load_tile_2d(
    uint32_t lds_addr, const void* gptr,
    uint32_t tile_d0 /*elems per row, =32*/, uint32_t tile_d1 /*rows*/,
    uint64_t row_stride_elems) {
  const uint64_t ga = (uint64_t)gptr;
  u32x4 g0;
  g0[0] = 1u;                                   // count=1, is_restore=0
  g0[1] = lds_addr;                             // lds_addr [63:32]
  g0[2] = (uint32_t)ga;                         // global_addr lo
  g0[3] = ((uint32_t)(ga >> 32) & 0x01FFFFFFu)  // global_addr [56:32]
          | (2u << 30);                         // type = 2 ("image")
  i32x8 g1;
  // data_size=1(2B) @[17:16]; pad_enable @[20]; pad_interval=3(16 DW) @[24:22];
  // pad_amount=3(4 DW) @[31:25]
  g1[0] = (int)((1u << 16) | (1u << 20) | (3u << 22) | (3u << 25));
  g1[1] = (int)((tile_d0 & 0xFFFFu) << 16);                       // tensor_dim0 lo
  g1[2] = (int)((tile_d0 >> 16) | ((tile_d1 & 0xFFFFu) << 16));   // td0 hi | tensor_dim1 lo
  g1[3] = (int)((tile_d1 >> 16) | (tile_d0 << 16));               // td1 hi | tile_dim0
  g1[4] = (int)(tile_d1 & 0xFFFFu);                               // tile_dim1 ; tile_dim2=0
  g1[5] = (int)(uint32_t)(row_stride_elems & 0xFFFFFFFFu);        // dim0_stride lo
  g1[6] = (int)(uint32_t)((row_stride_elems >> 32) & 0xFFFFu);    // dim0_stride hi
  g1[7] = 0;                                                      // dim1_stride hi
  const i32x4 z4 = {0, 0, 0, 0};
  const i32x8 z8 = {0, 0, 0, 0, 0, 0, 0, 0};
  __builtin_amdgcn_tensor_load_to_lds(g0, g1, z4, z4, z8, 0);
}

// ---------------------------------------------------------------------------
// Residual init: out[0:n) = x, out[n:2n) = y
// ---------------------------------------------------------------------------
__global__ __launch_bounds__(256)
void copy2_kernel(const float* __restrict__ x, const float* __restrict__ y,
                  float* __restrict__ out, int n) {
  int i = blockIdx.x * 256 + threadIdx.x;
  if (i < n) { out[i] = x[i]; out[(size_t)n + i] = y[i]; }
}

// ---------------------------------------------------------------------------
// Weight convert: f32 W[K x N] -> bf16 W^T [N x K]
// ---------------------------------------------------------------------------
__global__ __launch_bounds__(256)
void wconv_kernel(const float* __restrict__ W, bf16_t* __restrict__ Wt,
                  int K, int N) {
  const size_t idx = (size_t)blockIdx.x * 256 + threadIdx.x;
  if (idx >= (size_t)K * N) return;
  const int n = (int)(idx / K);
  const int k = (int)(idx % K);
  Wt[idx] = (bf16_t)W[(size_t)k * N + n];
}

// ---------------------------------------------------------------------------
// LayerNorm over D=768, one row per 256-thread block, bf16 output
// ---------------------------------------------------------------------------
__global__ __launch_bounds__(256)
void ln_bf16_kernel(const float* __restrict__ X, const float* __restrict__ gw,
                    const float* __restrict__ bw, bf16_t* __restrict__ Y) {
  const int row = blockIdx.x;
  const int tid = threadIdx.x;
  const float* x = X + (size_t)row * Dm;
  float v0 = x[tid], v1 = x[tid + 256], v2 = x[tid + 512];
  __shared__ float red[256];
  red[tid] = v0 + v1 + v2;
  __syncthreads();
  for (int s = 128; s > 0; s >>= 1) {
    if (tid < s) red[tid] += red[tid + s];
    __syncthreads();
  }
  const float mean = red[0] * (1.0f / Dm);
  __syncthreads();
  const float d0 = v0 - mean, d1 = v1 - mean, d2 = v2 - mean;
  red[tid] = d0 * d0 + d1 * d1 + d2 * d2;
  __syncthreads();
  for (int s = 128; s > 0; s >>= 1) {
    if (tid < s) red[tid] += red[tid + s];
    __syncthreads();
  }
  const float rstd = rsqrtf(red[0] * (1.0f / Dm) + 1e-5f);
  bf16_t* y = Y + (size_t)row * Dm;
  y[tid]       = (bf16_t)(d0 * rstd * gw[tid]       + bw[tid]);
  y[tid + 256] = (bf16_t)(d1 * rstd * gw[tid + 256] + bw[tid + 256]);
  y[tid + 512] = (bf16_t)(d2 * rstd * gw[tid + 512] + bw[tid + 512]);
}

// ---------------------------------------------------------------------------
// WMMA GEMM with TDM double-buffered tile pipeline.
// C[MxN] (+)= A[MxK] * B, with B given transposed: Bt[N x K].
// Block = 256 threads (8 waves), WG tile 128x64, wave tile 32x32, K-step 32.
// Wave 0 drives the Tensor Data Mover: tiles for K-step i+1 are DMA'd into
// the alternate LDS buffer while all 8 waves run WMMAs on K-step i.
// EPI: 0 = Cf = A*B ; 1 = Cf += A*B (+bias) ; 2 = Cb = gelu(A*B + bias)
// ---------------------------------------------------------------------------
template <int EPI>
__global__ __launch_bounds__(256)
void gemm_bf16_kernel(const bf16_t* __restrict__ A, const bf16_t* __restrict__ Bt,
                      float* __restrict__ Cf, bf16_t* __restrict__ Cb,
                      const float* __restrict__ bias, int M, int N, int K) {
  constexpr int LDS_PITCH = 40;   // 32 payload + 4-DWORD TDM pad per row
  __shared__ bf16_t sA[2][128 * LDS_PITCH];
  __shared__ bf16_t sB[2][64 * LDS_PITCH];
  const int tid  = threadIdx.x;
  const int wave = tid >> 5;
  const int lane = tid & 31;
  const int g    = lane >> 4;     // half-wave group
  const int l16  = lane & 15;
  const int wm   = (wave & 3) * 32;   // 4 wave-rows
  const int wn   = (wave >> 2) * 32;  // 2 wave-cols
  const int tileM = blockIdx.x * 128;
  const int tileN = blockIdx.y * 64;

  v8f acc[2][2];
  acc[0][0] = vzero8(); acc[0][1] = vzero8();
  acc[1][0] = vzero8(); acc[1][1] = vzero8();

  const int nK = K >> 5;   // K-steps of 32

  // prologue: DMA first tiles into buffer 0
  if (wave == 0) {
    tdm_load_tile_2d(lds_off(&sA[0][0]), A + (size_t)tileM * K, 32, 128, (uint64_t)K);
    tdm_load_tile_2d(lds_off(&sB[0][0]), Bt + (size_t)tileN * K, 32, 64, (uint64_t)K);
  }

  for (int s = 0; s < nK; ++s) {
    const int buf = s & 1;
    if (wave == 0) {
      if (s + 1 < nK) {
        const int k1 = (s + 1) * 32;
        tdm_load_tile_2d(lds_off(&sA[buf ^ 1][0]),
                         A + (size_t)tileM * K + k1, 32, 128, (uint64_t)K);
        tdm_load_tile_2d(lds_off(&sB[buf ^ 1][0]),
                         Bt + (size_t)tileN * K + k1, 32, 64, (uint64_t)K);
        __builtin_amdgcn_s_wait_tensorcnt(2);  // current-buffer pair complete
      } else {
        __builtin_amdgcn_s_wait_tensorcnt(0);
      }
    }
    __syncthreads();   // tiles of K-step s visible to all waves

    // --- fragments (ISA 7.12.2 layouts) ---
    v16bf af[2], bfr[2];
#pragma unroll
    for (int mi = 0; mi < 2; ++mi) {
      const bf16_t* p = &sA[buf][(wm + mi * 16 + l16) * LDS_PITCH];
      af[mi] = cat16(*(const v8bf*)(p + g * 8), *(const v8bf*)(p + 16 + g * 8));
    }
#pragma unroll
    for (int ni = 0; ni < 2; ++ni) {
      const bf16_t* p = &sB[buf][(wn + ni * 16 + l16) * LDS_PITCH];
      bfr[ni] = cat16(*(const v8bf*)(p + g * 8), *(const v8bf*)(p + 16 + g * 8));
    }
#pragma unroll
    for (int mi = 0; mi < 2; ++mi)
#pragma unroll
      for (int ni = 0; ni < 2; ++ni)
        acc[mi][ni] = wmma_bf16(af[mi], bfr[ni], acc[mi][ni]);
    __syncthreads();   // all reads of buf done before wave0 re-DMAs into it
  }

  // --- epilogue; C layout: vgpr r -> row r + 8*g, col = l16 ---
#pragma unroll
  for (int mi = 0; mi < 2; ++mi)
#pragma unroll
    for (int ni = 0; ni < 2; ++ni)
#pragma unroll
      for (int r = 0; r < 8; ++r) {
        const int row = tileM + wm + mi * 16 + r + 8 * g;
        const int col = tileN + wn + ni * 16 + l16;
        const size_t off = (size_t)row * N + col;
        float v = acc[mi][ni][r];
        if (EPI == 0) {
          Cf[off] = v;
        } else if (EPI == 1) {
          if (bias) v += bias[col];
          Cf[off] += v;
        } else {
          v += bias[col];
          const float ge = 0.5f * v * (1.0f + erff(v * 0.70710678118654752f));
          Cb[off] = (bf16_t)ge;
        }
      }
}

// ---------------------------------------------------------------------------
// QKV split, faithful to (d,3,h) channel interleave: chan = dd*36 + kk*12 + h.
// Q scaled by 1/sqrt(D/H)=1/8. V written transposed [B,H,DH,T].
// ---------------------------------------------------------------------------
__global__ __launch_bounds__(256)
void split_qkv_kernel(const float* __restrict__ qkv, bf16_t* __restrict__ Qo,
                      bf16_t* __restrict__ Ko, bf16_t* __restrict__ Vto) {
  const int idx = blockIdx.x * 256 + threadIdx.x;   // < (B*H) << 16
  const int dd = idx & 63;
  const int t  = (idx >> 6) & (Tseq - 1);
  const int bh = idx >> 16;                         // b*H + h
  const int h  = bh % Hh;
  const int b  = bh / Hh;
  const float* base = qkv + ((size_t)(b * Tseq + t)) * D3 + dd * (3 * Hh) + h;
  const float q = base[0] * 0.125f;
  const float k = base[Hh];
  const float v = base[2 * Hh];
  const size_t bhT = (size_t)bh * Tseq;
  Qo[(bhT + t) * DH + dd] = (bf16_t)q;
  Ko[(bhT + t) * DH + dd] = (bf16_t)k;
  Vto[((size_t)bh * DH + dd) * Tseq + t] = (bf16_t)v;
}

// ---------------------------------------------------------------------------
// Fused flash attention. Block = 128 (4 waves), each wave owns 16 query rows.
// Q [B,H,T,64] (pre-scaled), K [B,H,T,64], Vt [B,H,64,T]. Out bf16 [B*T, 768],
// column = h*64 + d (matches reference head-concat).
// ---------------------------------------------------------------------------
__global__ __launch_bounds__(128)
void attn_kernel(const bf16_t* __restrict__ Qm, const bf16_t* __restrict__ Km,
                 const bf16_t* __restrict__ Vtm, bf16_t* __restrict__ Outm) {
  constexpr int LP = 40;
  __shared__ bf16_t sP[4][16 * LP];
  const int bh   = blockIdx.x;          // b*H + h
  const int b    = bh / Hh;
  const int h    = bh % Hh;
  const int wave = threadIdx.x >> 5;
  const int lane = threadIdx.x & 31;
  const int g    = lane >> 4;
  const int l16  = lane & 15;
  const int qr0  = blockIdx.y * 64 + wave * 16;

  const bf16_t* qp = Qm  + ((size_t)bh * Tseq + qr0) * DH;
  const bf16_t* kp = Km  + (size_t)bh * Tseq * DH;
  const bf16_t* vp = Vtm + (size_t)bh * DH * Tseq;

  // Q A-fragments for head-dim halves 0..31 / 32..63
  v16bf qf[2];
#pragma unroll
  for (int s = 0; s < 2; ++s) {
    const bf16_t* p = qp + l16 * DH + s * 32 + g * 8;
    qf[s] = cat16(*(const v8bf*)p, *(const v8bf*)(p + 16));
  }

  v8f o[4];
  o[0] = vzero8(); o[1] = vzero8(); o[2] = vzero8(); o[3] = vzero8();
  float m[8], lsum[8];
#pragma unroll
  for (int r = 0; r < 8; ++r) { m[r] = -1e30f; lsum[r] = 0.f; }

  bf16_t* pw = sP[wave];

  for (int j = 0; j < Tseq; j += 32) {
    // ---- scores S = Q K^T for 32 keys (two 16-col subtiles) ----
    v8f s0 = vzero8(), s1 = vzero8();
#pragma unroll
    for (int s = 0; s < 2; ++s) {
      const bf16_t* kr0 = kp + (size_t)(j + l16) * DH + s * 32 + g * 8;
      v16bf b0 = cat16(*(const v8bf*)kr0, *(const v8bf*)(kr0 + 16));
      s0 = wmma_bf16(qf[s], b0, s0);
      const bf16_t* kr1 = kp + (size_t)(j + 16 + l16) * DH + s * 32 + g * 8;
      v16bf b1 = cat16(*(const v8bf*)kr1, *(const v8bf*)(kr1 + 16));
      s1 = wmma_bf16(qf[s], b1, s1);
    }

    // ---- online softmax (rows replicated across the 16 lanes of a half) ----
#pragma unroll
    for (int r = 0; r < 8; ++r) {
      float t = fmaxf(s0[r], s1[r]);
      t = fmaxf(t, __shfl_xor(t, 1, 32));
      t = fmaxf(t, __shfl_xor(t, 2, 32));
      t = fmaxf(t, __shfl_xor(t, 4, 32));
      t = fmaxf(t, __shfl_xor(t, 8, 32));
      const float mn = fmaxf(m[r], t);
      const float sc = __expf(m[r] - mn);
      const float p0 = __expf(s0[r] - mn);
      const float p1 = __expf(s1[r] - mn);
      float rs = p0 + p1;
      rs += __shfl_xor(rs, 1, 32);
      rs += __shfl_xor(rs, 2, 32);
      rs += __shfl_xor(rs, 4, 32);
      rs += __shfl_xor(rs, 8, 32);
      lsum[r] = lsum[r] * sc + rs;
      m[r] = mn;
      o[0][r] *= sc; o[1][r] *= sc; o[2][r] *= sc; o[3][r] *= sc;
      // stage P (C-layout -> LDS) for A-fragment reload
      pw[(r + 8 * g) * LP + l16]      = (bf16_t)p0;
      pw[(r + 8 * g) * LP + 16 + l16] = (bf16_t)p1;
    }
    // intra-wave LDS RAW ordering (split-counter ISA: DScnt)
    asm volatile("s_wait_dscnt 0" ::: "memory");

    const bf16_t* pr = pw + l16 * LP;
    v16bf pf = cat16(*(const v8bf*)(pr + g * 8), *(const v8bf*)(pr + 16 + g * 8));

    // ---- O += P * V (4 output subtiles of head dim) ----
#pragma unroll
    for (int n = 0; n < 4; ++n) {
      const bf16_t* vr = vp + (size_t)(n * 16 + l16) * Tseq + j + g * 8;
      v16bf vf = cat16(*(const v8bf*)vr, *(const v8bf*)(vr + 16));
      o[n] = wmma_bf16(pf, vf, o[n]);
    }
  }

  // ---- normalize and write bf16 (concat heads) ----
#pragma unroll
  for (int n = 0; n < 4; ++n)
#pragma unroll
    for (int r = 0; r < 8; ++r) {
      const int row = qr0 + r + 8 * g;
      const int col = h * DH + n * 16 + l16;
      Outm[((size_t)b * Tseq + row) * Dm + col] = (bf16_t)(o[n][r] / lsum[r]);
    }
}

// ---------------------------------------------------------------------------
// Host orchestration
// ---------------------------------------------------------------------------
extern "C" void kernel_launch(void* const* d_in, const int* in_sizes, int n_in,
                              void* d_out, int out_size, void* d_ws, size_t ws_size,
                              hipStream_t stream) {
  (void)in_sizes; (void)n_in; (void)out_size; (void)ws_size;
  const float* x  = (const float*)d_in[0];
  const float* y  = (const float*)d_in[1];
  const float* w_qkv_a  = (const float*)d_in[2];
  const float* w_out_a  = (const float*)d_in[3];
  const float* w_qkv_b  = (const float*)d_in[4];
  const float* w_out_b  = (const float*)d_in[5];
  const float* cw_qkv_a = (const float*)d_in[6];
  const float* cw_out_a = (const float*)d_in[7];
  const float* cw_qkv_b = (const float*)d_in[8];
  const float* cw_out_b = (const float*)d_in[9];
  const float* mlp_a_w1 = (const float*)d_in[10];
  const float* mlp_a_b1 = (const float*)d_in[11];
  const float* mlp_a_w2 = (const float*)d_in[12];
  const float* mlp_a_b2 = (const float*)d_in[13];
  const float* mlp_b_w1 = (const float*)d_in[14];
  const float* mlp_b_b1 = (const float*)d_in[15];
  const float* mlp_b_w2 = (const float*)d_in[16];
  const float* mlp_b_b2 = (const float*)d_in[17];
  const float* ln_g[6], *ln_b[6];
  for (int i = 0; i < 6; ++i) { ln_g[i] = (const float*)d_in[18 + 2 * i];
                                ln_b[i] = (const float*)d_in[19 + 2 * i]; }

  float* out_x = (float*)d_out;
  float* out_y = out_x + (size_t)BT * Dm;

  // --- workspace carve-up ---
  uint8_t* wsp = (uint8_t*)d_ws;
  auto alloc = [&](size_t bytes) -> void* {
    void* p = wsp;
    wsp += (bytes + 255) & ~(size_t)255;
    return p;
  };
  bf16_t* wqaT = (bf16_t*)alloc((size_t)D3 * Dm * 2);    // [2304 x 768]
  bf16_t* wqbT = (bf16_t*)alloc((size_t)D3 * Dm * 2);
  bf16_t* cqaT = (bf16_t*)alloc((size_t)D3 * Dm * 2);
  bf16_t* cqbT = (bf16_t*)alloc((size_t)D3 * Dm * 2);
  bf16_t* woaT = (bf16_t*)alloc((size_t)Dm * Dm * 2);
  bf16_t* wobT = (bf16_t*)alloc((size_t)Dm * Dm * 2);
  bf16_t* coaT = (bf16_t*)alloc((size_t)Dm * Dm * 2);
  bf16_t* cobT = (bf16_t*)alloc((size_t)Dm * Dm * 2);
  bf16_t* w1aT = (bf16_t*)alloc((size_t)MLPD * Dm * 2);  // [3072 x 768]
  bf16_t* w1bT = (bf16_t*)alloc((size_t)MLPD * Dm * 2);
  bf16_t* w2aT = (bf16_t*)alloc((size_t)Dm * MLPD * 2);  // [768 x 3072]
  bf16_t* w2bT = (bf16_t*)alloc((size_t)Dm * MLPD * 2);
  bf16_t* lnx  = (bf16_t*)alloc((size_t)BT * Dm * 2);
  bf16_t* lny  = (bf16_t*)alloc((size_t)BT * Dm * 2);
  float*  qkvx = (float*)alloc((size_t)BT * D3 * 4);     // reused as MLP hidden
  float*  qkvy = (float*)alloc((size_t)BT * D3 * 4);
  bf16_t* Qx   = (bf16_t*)alloc((size_t)BT * DH * Hh * 2);
  bf16_t* Kx   = (bf16_t*)alloc((size_t)BT * DH * Hh * 2);
  bf16_t* Vtx  = (bf16_t*)alloc((size_t)BT * DH * Hh * 2);
  bf16_t* Qy   = (bf16_t*)alloc((size_t)BT * DH * Hh * 2);
  bf16_t* Ky   = (bf16_t*)alloc((size_t)BT * DH * Hh * 2);
  bf16_t* Vty  = (bf16_t*)alloc((size_t)BT * DH * Hh * 2);
  bf16_t* attx = (bf16_t*)alloc((size_t)BT * Dm * 2);
  bf16_t* atty = (bf16_t*)alloc((size_t)BT * Dm * 2);
  bf16_t* hx   = (bf16_t*)qkvx;   // alias: MLP hidden (25.2MB <= 37.7MB)
  bf16_t* hy   = (bf16_t*)qkvy;

  const int nTok = BT * Dm;   // 3,145,728

  // residual init
  copy2_kernel<<<nTok / 256, 256, 0, stream>>>(x, y, out_x, nTok);

  // weight conversion (f32 [KxN] -> bf16 [NxK])
  auto conv = [&](const float* W, bf16_t* Wt, int K, int N) {
    const size_t e = (size_t)K * N;
    wconv_kernel<<<(unsigned)((e + 255) / 256), 256, 0, stream>>>(W, Wt, K, N);
  };
  conv(w_qkv_a,  wqaT, Dm, D3);   conv(w_qkv_b,  wqbT, Dm, D3);
  conv(cw_qkv_a, cqaT, Dm, D3);   conv(cw_qkv_b, cqbT, Dm, D3);
  conv(w_out_a,  woaT, Dm, Dm);   conv(w_out_b,  wobT, Dm, Dm);
  conv(cw_out_a, coaT, Dm, Dm);   conv(cw_out_b, cobT, Dm, Dm);
  conv(mlp_a_w1, w1aT, Dm, MLPD); conv(mlp_b_w1, w1bT, Dm, MLPD);
  conv(mlp_a_w2, w2aT, MLPD, Dm); conv(mlp_b_w2, w2bT, MLPD, Dm);

  auto gemm0 = [&](const bf16_t* A, const bf16_t* Bt, float* C, int M, int N, int K) {
    gemm_bf16_kernel<0><<<dim3(M / 128, N / 64), 256, 0, stream>>>(
        A, Bt, C, nullptr, nullptr, M, N, K);
  };
  auto gemm1 = [&](const bf16_t* A, const bf16_t* Bt, float* C,
                   const float* bias, int M, int N, int K) {
    gemm_bf16_kernel<1><<<dim3(M / 128, N / 64), 256, 0, stream>>>(
        A, Bt, C, nullptr, bias, M, N, K);
  };
  auto gemm2 = [&](const bf16_t* A, const bf16_t* Bt, bf16_t* C,
                   const float* bias, int M, int N, int K) {
    gemm_bf16_kernel<2><<<dim3(M / 128, N / 64), 256, 0, stream>>>(
        A, Bt, nullptr, C, bias, M, N, K);
  };
  auto ln = [&](const float* X, int li, bf16_t* Y) {
    ln_bf16_kernel<<<BT, 256, 0, stream>>>(X, ln_g[li], ln_b[li], Y);
  };
  auto split = [&](const float* qkv, bf16_t* Q, bf16_t* K, bf16_t* Vt) {
    split_qkv_kernel<<<(Bsz * Hh << 16) / 256, 256, 0, stream>>>(qkv, Q, K, Vt);
  };
  auto attn = [&](const bf16_t* Q, const bf16_t* K, const bf16_t* Vt, bf16_t* O) {
    attn_kernel<<<dim3(Bsz * Hh, Tseq / 64), 128, 0, stream>>>(Q, K, Vt, O);
  };

  // ===== stage 1: self-attention =====
  ln(out_x, 0, lnx);                       // ln1
  ln(out_y, 3, lny);                       // ln4
  gemm0(lnx, wqaT, qkvx, BT, D3, Dm);
  gemm0(lny, wqbT, qkvy, BT, D3, Dm);
  split(qkvx, Qx, Kx, Vtx);
  split(qkvy, Qy, Ky, Vty);
  attn(Qx, Kx, Vtx, attx);
  attn(Qy, Ky, Vty, atty);
  gemm1(attx, woaT, out_x, nullptr, BT, Dm, Dm);
  gemm1(atty, wobT, out_y, nullptr, BT, Dm, Dm);

  // ===== stage 2: cross-attention =====
  ln(out_x, 1, lnx);                       // ln2
  ln(out_y, 4, lny);                       // ln5
  gemm0(lnx, cqaT, qkvx, BT, D3, Dm);      // q1,k1,v1 from x
  gemm0(lny, cqbT, qkvy, BT, D3, Dm);      // q2,k2,v2 from y
  split(qkvx, Qx, Kx, Vtx);
  split(qkvy, Qy, Ky, Vty);
  attn(Qy, Kx, Vtx, attx);                 // attend(q2, k1, v1)
  attn(Qx, Ky, Vty, atty);                 // attend(q1, k2, v2)
  gemm1(attx, coaT, out_x, nullptr, BT, Dm, Dm);
  gemm1(atty, cobT, out_y, nullptr, BT, Dm, Dm);

  // ===== stage 3: MLP =====
  ln(out_x, 2, lnx);                       // ln3
  ln(out_y, 5, lny);                       // ln6
  gemm2(lnx, w1aT, hx, mlp_a_b1, BT, MLPD, Dm);   // gelu(fc1)
  gemm2(lny, w1bT, hy, mlp_b_b1, BT, MLPD, Dm);
  gemm1(hx, w2aT, out_x, mlp_a_b2, BT, Dm, MLPD); // residual += fc2 + b2
  gemm1(hy, w2bT, out_y, mlp_b_b2, BT, Dm, MLPD);
}
